// RNN_24902220382915
// MI455X (gfx1250) — matrix operand order repaired and checked by reading, compiled, not run on hardware
//
#include <hip/hip_runtime.h>

// ---------------------------------------------------------------------------
// RNN forward for MI455X (gfx1250, wave32, WMMA).
//   B=64, S=512, D_IN=512, D_H=1024
// Phase 1: xp = x @ w_xh + (b_xh + b_hh)   (bf16 WMMA, 16x64 tile/wave,
//          2-stage software pipeline: loads of stage k+1 overlap WMMAs of k)
// Phase 2: persistent kernel, 512 steps of h = tanh(xp_t + h @ w_hh)
//          - w_hh fragments preloaded in registers (read once, not 512x)
//          - split-K over 8 waves + LDS reduction (short dependence chain)
//          - device-wide barrier between steps (no launch overhead)
// ---------------------------------------------------------------------------

typedef __bf16 bf16_t;
typedef __attribute__((ext_vector_type(16))) __bf16 v16bf;
typedef __attribute__((ext_vector_type(8)))  __bf16 v8bf;
typedef __attribute__((ext_vector_type(8)))  float  v8f;
typedef __attribute__((ext_vector_type(4)))  float  v4f;

#define BB   64
#define SS   512
#define DIN  512
#define DH   1024
#define NBLK 256                                   // blocks in scan kernel

// B fragment: elems 0-7 = K kb+8h.., elems 8-15 = K kb+16+8h..
__device__ __forceinline__ v16bf load_b_frag(const bf16_t* p) {
  v8bf b0 = *(const v8bf*)p;
  v8bf b1 = *(const v8bf*)(p + 16);
  v16bf b;
#pragma unroll
  for (int i = 0; i < 8; ++i) { b[i] = b0[i]; b[8 + i] = b1[i]; }
  return b;
}

// A fragment (bf16 source): elems 0-7 = K kb+8h.., elems 8-15 = K kb+16+8h..
__device__ __forceinline__ v16bf load_a_frag_bf(const bf16_t* p) {
  v8bf a0 = *(const v8bf*)p;
  v8bf a1 = *(const v8bf*)(p + 16);
  v16bf a;
#pragma unroll
  for (int i = 0; i < 8; ++i) { a[i] = a0[i]; a[8 + i] = a1[i]; }
  return a;
}

// A fragment from f32 source (converted in-register).
__device__ __forceinline__ v16bf load_a_frag_f32(const float* p) {
  v4f f0 = *(const v4f*)p;
  v4f f1 = *(const v4f*)(p + 4);
  v4f f2 = *(const v4f*)(p + 16);
  v4f f3 = *(const v4f*)(p + 20);
  v16bf a;
#pragma unroll
  for (int i = 0; i < 4; ++i) {
    a[i]      = (bf16_t)f0[i];
    a[4 + i]  = (bf16_t)f1[i];
    a[8 + i]  = (bf16_t)f2[i];
    a[12 + i] = (bf16_t)f3[i];
  }
  return a;
}

// ---------------------------------------------------------------------------
// Prep: transpose weights to [N,K] bf16, fuse biases, broadcast h0, reset sync.
// ---------------------------------------------------------------------------
__global__ __launch_bounds__(256) void prep_kernel(
    const float* __restrict__ w_xh, const float* __restrict__ b_xh,
    const float* __restrict__ w_hh, const float* __restrict__ b_hh,
    const float* __restrict__ h0,
    bf16_t* __restrict__ wxhT, bf16_t* __restrict__ whhT,
    float* __restrict__ bias, bf16_t* __restrict__ hbuf,
    int* __restrict__ sync_cnt)
{
  int idx = blockIdx.x * 256 + threadIdx.x;
  if (idx == 0) *sync_cnt = 0;
  if (idx < DH * DH) {                       // w_hh: [K=DH, N=DH] -> whhT[N][K]
    int n = idx / DH, k = idx % DH;
    whhT[(size_t)n * DH + k] = (bf16_t)w_hh[(size_t)k * DH + n];
  }
  if (idx < DH * DIN) {                      // w_xh: [K=DIN, N=DH] -> wxhT[N][K]
    int n = idx / DIN, k = idx % DIN;
    wxhT[(size_t)n * DIN + k] = (bf16_t)w_xh[(size_t)k * DH + n];
  }
  if (idx < DH) bias[idx] = b_xh[idx] + b_hh[idx];
  if (idx < BB * DH) hbuf[idx] = (bf16_t)h0[idx & (DH - 1)];
}

// ---------------------------------------------------------------------------
// Phase 1 GEMM: xp[M=32768, N=1024] = x[M,512] * wxhT^T + bias
// 16(M) x 64(N) register tile per wave. Two fragment sets ping-pong: while
// WMMAs consume set 0, set 1's loads are in flight (and vice versa) -- no
// register moves, no WMMA->VALU hazard NOPs.
// ---------------------------------------------------------------------------
__global__ __launch_bounds__(256) void xproj_kernel(
    const float* __restrict__ x, const bf16_t* __restrict__ wxhT,
    const float* __restrict__ bias, float* __restrict__ xp)
{
  const int lane = threadIdx.x & 31;
  const int wave = threadIdx.x >> 5;
  const int tile = blockIdx.x * 8 + wave;    // 2048 * 16 = 32768 wave-tiles
  const int tN   = tile & 15;                // 16 N-supertiles (64 cols each)
  const int tM   = tile >> 4;                // 2048 M-tiles
  const int half = lane >> 4;
  const int lm   = lane & 15;

  const float*  arow  = x    + (size_t)(tM * 16 + lm) * DIN + 8 * half;
  const bf16_t* bbase = wxhT + (size_t)(tN * 64 + lm) * DIN + 8 * half;

  v8f c[4] = {{}, {}, {}, {}};

  // Prologue: stage-0 fragments for kb = 0.
  v16bf a0 = load_a_frag_f32(arow);
  v16bf b0[4];
#pragma unroll
  for (int j = 0; j < 4; ++j) b0[j] = load_b_frag(bbase + (size_t)j * 16 * DIN);

  for (int kb = 0; kb < DIN; kb += 64) {
    // Issue stage-1 loads (kb+32), then run stage-0 WMMAs.
    v16bf a1 = load_a_frag_f32(arow + kb + 32);
    v16bf b1[4];
#pragma unroll
    for (int j = 0; j < 4; ++j)
      b1[j] = load_b_frag(bbase + (size_t)j * 16 * DIN + kb + 32);
#pragma unroll
    for (int j = 0; j < 4; ++j)
      c[j] = __builtin_amdgcn_wmma_f32_16x16x32_bf16(
          false, a0, false, b0[j], (short)0, c[j], false, false);

    // Issue stage-0 loads for kb+64 (clamped on last iter), run stage-1 WMMAs.
    const int kn = (kb + 64 < DIN) ? kb + 64 : kb;   // uniform clamp, L2-hot
    a0 = load_a_frag_f32(arow + kn);
#pragma unroll
    for (int j = 0; j < 4; ++j)
      b0[j] = load_b_frag(bbase + (size_t)j * 16 * DIN + kn);
#pragma unroll
    for (int j = 0; j < 4; ++j)
      c[j] = __builtin_amdgcn_wmma_f32_16x16x32_bf16(
          false, a1, false, b1[j], (short)0, c[j], false, false);
  }

#pragma unroll
  for (int j = 0; j < 4; ++j) {
    const int col = tN * 64 + j * 16 + lm;
    const float bv = bias[col];
#pragma unroll
    for (int r = 0; r < 8; ++r) {            // C/D: VGPR r -> row r + 8*half
      int row = tM * 16 + r + 8 * half;
      xp[(size_t)row * DH + col] = c[j][r] + bv;
    }
  }
}

// ---------------------------------------------------------------------------
// Phase 2: persistent scan kernel. One block per 16x16 output tile; wave w
// owns K slice [w*128, w*128+128). Its 4 w_hh fragments are preloaded into
// registers ONCE. Each step: load h frags -> 4 WMMAs -> LDS reduce ->
// tanh/store -> device-wide barrier.
// ---------------------------------------------------------------------------
__global__ __launch_bounds__(256) void rnn_scan_kernel(
    bf16_t* __restrict__ hA, bf16_t* __restrict__ hB,
    const bf16_t* __restrict__ whhT, float* __restrict__ hs,
    int* __restrict__ sync_cnt)
{
  __shared__ float red[8 * 256];             // 8 partial 16x16 f32 tiles

  const int lane = threadIdx.x & 31;
  const int wave = threadIdx.x >> 5;
  const int tN   = blockIdx.x & 63;          // 64 N-tiles
  const int tM   = blockIdx.x >> 6;          // 4 M-tiles (batch groups)
  const int half = lane >> 4;
  const int lm   = lane & 15;

  // Loop-invariant B fragments: this wave's 16 cols x 128-wide K slice.
  const bf16_t* brow = whhT + (size_t)(tN * 16 + lm) * DH + wave * 128 + 8 * half;
  v16bf bfrag[4];
#pragma unroll
  for (int kk = 0; kk < 4; ++kk)
    bfrag[kk] = load_b_frag(brow + kk * 32);

  const bf16_t* arow0 = hA + (size_t)(tM * 16 + lm) * DH + wave * 128 + 8 * half;
  const bf16_t* arow1 = hB + (size_t)(tM * 16 + lm) * DH + wave * 128 + 8 * half;

  // Epilogue assignment: one thread per tile element.
  const int m = threadIdx.x >> 4, n = threadIdx.x & 15;
  const int b_out = tM * 16 + m, col_out = tN * 16 + n;
  bf16_t* hn_even = hB + (size_t)b_out * DH + col_out;   // write when t even
  bf16_t* hn_odd  = hA + (size_t)b_out * DH + col_out;   // write when t odd

  for (int t = 0; t < SS; ++t) {
    const bf16_t* arow = (t & 1) ? arow1 : arow0;

    v8f c = {};
#pragma unroll
    for (int kk = 0; kk < 4; ++kk) {
      v16bf a = load_a_frag_bf(arow + kk * 32);
      c = __builtin_amdgcn_wmma_f32_16x16x32_bf16(
          false, a, false, bfrag[kk], (short)0, c, false, false);
    }

    // Partial tile -> LDS: element (m, n) at m*16 + n
#pragma unroll
    for (int r = 0; r < 8; ++r)
      red[wave * 256 + (r + 8 * half) * 16 + lm] = c[r];
    __syncthreads();

    float acc = 0.f;
#pragma unroll
    for (int w = 0; w < 8; ++w) acc += red[w * 256 + threadIdx.x];

    const size_t off = ((size_t)b_out * SS + t) * DH + col_out;
    const float v = tanhf(acc + hs[off]);
    hs[off] = v;                             // hidden_tensors[b, t, :]
    *((t & 1) ? hn_odd : hn_even) = (bf16_t)v;
    if (t == SS - 1)                         // h_last tail of d_out
      hs[(size_t)BB * SS * DH + (size_t)b_out * DH + col_out] = v;

    // Device-wide barrier: make h visible, then monotonic-counter rendezvous.
    __threadfence();
    __syncthreads();                         // also guards LDS reuse
    if (threadIdx.x == 0) {
      __hip_atomic_fetch_add(sync_cnt, 1, __ATOMIC_ACQ_REL,
                             __HIP_MEMORY_SCOPE_AGENT);
      const int target = NBLK * (t + 1);
      while (__hip_atomic_load(sync_cnt, __ATOMIC_ACQUIRE,
                               __HIP_MEMORY_SCOPE_AGENT) < target)
        __builtin_amdgcn_s_sleep(1);
    }
    __syncthreads();
  }
}

// ---------------------------------------------------------------------------
extern "C" void kernel_launch(void* const* d_in, const int* in_sizes, int n_in,
                              void* d_out, int out_size, void* d_ws, size_t ws_size,
                              hipStream_t stream) {
  (void)in_sizes; (void)n_in; (void)out_size; (void)ws_size;

  const float* x    = (const float*)d_in[0];
  const float* w_xh = (const float*)d_in[1];
  const float* b_xh = (const float*)d_in[2];
  const float* w_hh = (const float*)d_in[3];
  const float* b_hh = (const float*)d_in[4];
  const float* h0   = (const float*)d_in[5];
  float* out = (float*)d_out;

  // Workspace layout (~3.3 MB)
  char* ws = (char*)d_ws;
  bf16_t* wxhT = (bf16_t*)ws;                            // 1 MB  [DH][DIN]
  bf16_t* whhT = (bf16_t*)(ws + (1u << 20));             // 2 MB  [DH][DH]
  float*  bias = (float*)(ws + (3u << 20));              // 4 KB
  bf16_t* hA   = (bf16_t*)(ws + (3u << 20) + 4096);      // 128 KB [BB][DH]
  bf16_t* hBv  = hA + BB * DH;                           // 128 KB
  int*    cnt  = (int*)(hBv + BB * DH);                  // grid-sync counter

  // Prep: transposed bf16 weights, fused bias, bf16 h0, sync counter reset
  prep_kernel<<<(DH * DH) / 256, 256, 0, stream>>>(
      w_xh, b_xh, w_hh, b_hh, h0, wxhT, whhT, bias, hA, cnt);

  // Phase 1: xp -> hidden region of d_out (in-place staging)
  xproj_kernel<<<(2048 * 16) / 8, 256, 0, stream>>>(x, wxhT, bias, out);

  // Phase 2: one persistent launch runs all 512 steps + tail
  rnn_scan_kernel<<<NBLK, 256, 0, stream>>>(hA, hBv, whhT, out, cnt);
}
